// DMPNN_77756087927648
// MI455X (gfx1250) — matrix operand order repaired and checked by reading, not compiled, not run on hardware
//
#include <hip/hip_runtime.h>
#include <hip/hip_bf16.h>
#include <math.h>

typedef __attribute__((ext_vector_type(16))) _Float16 v16h;
typedef __attribute__((ext_vector_type(8)))  _Float16 v8h;
typedef __attribute__((ext_vector_type(8)))  float    v8f;

#define BB 64
#define NN 512
#define BN 32768          // B*N
#define EE 524288
#define DH 200
#define KPAD 224          // DH padded to mult of 32 (WMMA K for att)
#define DPAD 208          // DH padded to mult of 16 (WMMA N)
#define XKP 416           // 2*DH padded to mult of 32 (WMMA K for down-proj)
#define NT 13             // number of 16-wide N tiles covering DPAD
#define T_SCALE 0.07071067811865475f

// ---------------- wave32 reductions ----------------
__device__ __forceinline__ float wred_sum(float v) {
#pragma unroll
    for (int m = 16; m >= 1; m >>= 1) v += __shfl_xor(v, m, 32);
    return v;
}
__device__ __forceinline__ float wred_max(float v) {
#pragma unroll
    for (int m = 16; m >= 1; m >>= 1) v = fmaxf(v, __shfl_xor(v, m, 32));
    return v;
}

__device__ __forceinline__ v16h make_a(v8h lo, v8h hi) {
    return __builtin_shufflevector(lo, hi, 0, 1, 2, 3, 4, 5, 6, 7,
                                   8, 9, 10, 11, 12, 13, 14, 15);
}

// ---------------- 1) atom embed + layernorm ----------------
__global__ void atom_kernel(const float* __restrict__ atom,
                            const float* __restrict__ W_atom,
                            const float* __restrict__ ln_g,
                            const float* __restrict__ ln_b,
                            float* __restrict__ h) {
    int row  = blockIdx.x * 8 + (threadIdx.x >> 5);
    int lane = threadIdx.x & 31;
    const float* ar = atom + (size_t)row * 70;
    float vals[7];
    float sum = 0.f, sumsq = 0.f;
#pragma unroll
    for (int t = 0; t < 7; ++t) {
        int d = lane + 32 * t;
        float v = 0.f;
        if (d < DH) {
            float a = 0.f;
            for (int k = 0; k < 70; ++k) a += ar[k] * W_atom[k * DH + d];
            v = fmaxf(a, 0.f);
            sum += v; sumsq += v * v;
        }
        vals[t] = v;
    }
    sum   = wred_sum(sum);
    sumsq = wred_sum(sumsq);
    float mean = sum / (float)DH;
    float var  = sumsq / (float)DH - mean * mean;
    float rstd = rsqrtf(var + 1e-5f);
    float* hr = h + (size_t)row * DH;
#pragma unroll
    for (int t = 0; t < 7; ++t) {
        int d = lane + 32 * t;
        if (d < DH) hr[d] = ln_g[d] * (vals[t] - mean) * rstd + ln_b[d];
    }
}

// ---------------- 2) per-edge message + scatter ----------------
__global__ void edge_kernel(const float* __restrict__ efeat,
                            const float* __restrict__ coords,
                            const int* __restrict__ src,
                            const int* __restrict__ dst,
                            const float* __restrict__ W_edge,
                            const float* __restrict__ W_rbf,
                            const float* __restrict__ b_rbf,
                            const float* __restrict__ ln_g,
                            const float* __restrict__ ln_b,
                            float* __restrict__ h,
                            int* __restrict__ deg) {
    int e    = blockIdx.x * 8 + (threadIdx.x >> 5);
    int lane = threadIdx.x & 31;
    const float* ef = efeat + (size_t)e * 14;
    int s = src[e], d = dst[e];
    float dx = coords[s * 3 + 0] - coords[d * 3 + 0];
    float dy = coords[s * 3 + 1] - coords[d * 3 + 1];
    float dz = coords[s * 3 + 2] - coords[d * 3 + 2];
    float dist = sqrtf(dx * dx + dy * dy + dz * dz + 1e-12f);
    float rbf[16];
#pragma unroll
    for (int k = 0; k < 16; ++k) {
        float mu = 5.0f * (float)k / 15.0f;
        float u  = (dist - mu) / 0.3125f;
        rbf[k] = __expf(-u * u);
    }
    float vals[7];
    float sum = 0.f, sumsq = 0.f;
#pragma unroll
    for (int t = 0; t < 7; ++t) {
        int j = lane + 32 * t;
        float v = 0.f;
        if (j < 100) {
            float a = 0.f;
            for (int k = 0; k < 14; ++k) a += ef[k] * W_edge[k * 100 + j];
            v = fmaxf(a, 0.f);
        } else if (j < DH) {
            int jj = j - 100;
            float a = b_rbf[jj];
            for (int k = 0; k < 16; ++k) a += rbf[k] * W_rbf[k * 100 + jj];
            v = fmaxf(a, 0.f);
        }
        vals[t] = v;
        if (j < DH) { sum += v; sumsq += v * v; }
    }
    sum   = wred_sum(sum);
    sumsq = wred_sum(sumsq);
    float mean = sum / (float)DH;
    float var  = sumsq / (float)DH - mean * mean;
    float rstd = rsqrtf(var + 1e-5f);
    float* hr = h + (size_t)d * DH;
#pragma unroll
    for (int t = 0; t < 7; ++t) {
        int j = lane + 32 * t;
        if (j < DH) {
            float o = ln_g[j] * (vals[t] - mean) * rstd + ln_b[j];
            atomicAdd(&hr[j], o);
        }
    }
    if (lane == 0) atomicAdd(&deg[d], 1);
}

// ---------------- 3) pack h -> f16 padded row-major + f16 transposed ----------------
__global__ void pack_kernel(const float* __restrict__ h,
                            _Float16* __restrict__ hpad,
                            _Float16* __restrict__ hT) {
    size_t idx = (size_t)blockIdx.x * 256 + threadIdx.x;   // over BN*KPAD
    int row = (int)(idx / KPAD);
    int d   = (int)(idx % KPAD);
    float v = (d < DH) ? h[(size_t)row * DH + d] : 0.f;
    hpad[(size_t)row * KPAD + d] = (_Float16)v;
    if (d < DPAD) {
        int b = row >> 9, n = row & 511;
        hT[((size_t)b * DPAD + d) * NN + n] = (_Float16)v;
    }
}

// ---------------- 4) attention logits: att = T_SCALE * h1 @ h2^T (WMMA) ----------------
__global__ void att_kernel(const _Float16* __restrict__ h1pad,
                           const _Float16* __restrict__ h2pad,
                           float* __restrict__ att) {
    int b    = blockIdx.z;
    int bm   = blockIdx.y * 64;
    int bn   = blockIdx.x * 64;
    int wave = threadIdx.x >> 5;
    int lane = threadIdx.x & 31;
    int half = lane >> 4;
    int lrow = lane & 15;
    int tr   = wave & 3;
    int tc0  = (wave >> 2) * 2;

    const _Float16* h1b = h1pad + (size_t)b * NN * KPAD;
    const _Float16* h2b = h2pad + (size_t)b * NN * KPAD;

    int m_g = bm + tr * 16 + lrow;
    const _Float16* arow  = h1b + (size_t)m_g * KPAD;
    const _Float16* brow0 = h2b + (size_t)(bn + tc0 * 16 + lrow) * KPAD;
    const _Float16* brow1 = h2b + (size_t)(bn + (tc0 + 1) * 16 + lrow) * KPAD;

    v8f acc0 = {}, acc1 = {};
    for (int k0 = 0; k0 < KPAD; k0 += 32) {
        int ka = k0 + half * 16;
        v16h a  = make_a(*(const v8h*)(arow + ka), *(const v8h*)(arow + ka + 8));
        v16h b0 = *(const v16h*)(brow0 + k0 + half * 16);
        v16h b1 = *(const v16h*)(brow1 + k0 + half * 16);
        acc0 = __builtin_amdgcn_wmma_f32_16x16x32_f16(false, a, false, b0,
                                                      (short)0, acc0, false, false);
        acc1 = __builtin_amdgcn_wmma_f32_16x16x32_f16(false, a, false, b1,
                                                      (short)0, acc1, false, false);
    }
    float* out = att + (size_t)b * NN * NN;
#pragma unroll
    for (int r = 0; r < 8; ++r) {
        int mr = bm + tr * 16 + r + 8 * half;
        out[(size_t)mr * NN + bn + tc0 * 16 + lrow]       = acc0[r] * T_SCALE;
        out[(size_t)mr * NN + bn + (tc0 + 1) * 16 + lrow] = acc1[r] * T_SCALE;
    }
}

// ---------------- 5) softmax stats ----------------
__global__ void rowstats_kernel(const float* __restrict__ att,
                                float* __restrict__ rmax, float* __restrict__ rsum) {
    int row  = blockIdx.x * 8 + (threadIdx.x >> 5);   // b*512 + i
    int lane = threadIdx.x & 31;
    const float* a = att + (size_t)row * NN;
    float v[16], m = -1e30f;
#pragma unroll
    for (int t = 0; t < 16; ++t) { v[t] = a[lane + 32 * t]; m = fmaxf(m, v[t]); }
    m = wred_max(m);
    float s = 0.f;
#pragma unroll
    for (int t = 0; t < 16; ++t) s += __expf(v[t] - m);
    s = wred_sum(s);
    if (lane == 0) { rmax[row] = m; rsum[row] = s; }
}

__global__ void colstats_kernel(const float* __restrict__ att,
                                float* __restrict__ cmax, float* __restrict__ csum) {
    int gid  = blockIdx.x * 8 + (threadIdx.x >> 5);   // b*512 + j
    int lane = threadIdx.x & 31;
    int b = gid >> 9, j = gid & 511;
    const float* a = att + (size_t)b * NN * NN + j;
    float v[16], m = -1e30f;
#pragma unroll
    for (int t = 0; t < 16; ++t) { v[t] = a[(size_t)(lane + 32 * t) * NN]; m = fmaxf(m, v[t]); }
    m = wred_max(m);
    float s = 0.f;
#pragma unroll
    for (int t = 0; t < 16; ++t) s += __expf(v[t] - m);
    s = wred_sum(s);
    if (lane == 0) { cmax[gid] = m; csum[gid] = s; }
}

// ---------------- 6) c1 = softmax_row(att) @ h2 ----------------
// register-blocked: one wave per (b, m-tile), 13 accumulator tiles,
// A-fragment (exp fused) built once per k-step and reused 13x.
__global__ void c1_kernel(const float* __restrict__ att,
                          const float* __restrict__ rmax, const float* __restrict__ rsum,
                          const _Float16* __restrict__ h2T,
                          float* __restrict__ c1) {
    int wid  = blockIdx.x * 8 + (threadIdx.x >> 5);   // b*32 + mt
    int lane = threadIdx.x & 31;
    int half = lane >> 4, lrow = lane & 15;
    int b = wid >> 5, mt = wid & 31;

    int m_g = mt * 16 + lrow;
    float rmx = rmax[b * NN + m_g];
    const float*    arow  = att + ((size_t)b * NN + m_g) * NN;
    const _Float16* bbase = h2T + ((size_t)b * DPAD + lrow) * NN;

    v8f acc[NT];
#pragma unroll
    for (int t = 0; t < NT; ++t) acc[t] = (v8f){};

    for (int k0 = 0; k0 < NN; k0 += 32) {
        int ka = k0 + half * 16;
        v16h a;
#pragma unroll
        for (int i = 0; i < 8; ++i) a[i]     = (_Float16)__expf(arow[ka + i] - rmx);
#pragma unroll
        for (int i = 0; i < 8; ++i) a[8 + i] = (_Float16)__expf(arow[ka + 8 + i] - rmx);
#pragma unroll
        for (int dt = 0; dt < NT; ++dt) {
            v16h bv = *(const v16h*)(bbase + (size_t)dt * 16 * NN + k0 + half * 16);
            acc[dt] = __builtin_amdgcn_wmma_f32_16x16x32_f16(false, a, false, bv,
                                                             (short)0, acc[dt], false, false);
        }
    }
#pragma unroll
    for (int r = 0; r < 8; ++r) {
        int mrow = mt * 16 + r + 8 * half;
        float inv = 1.0f / rsum[b * NN + mrow];
        float* crow = c1 + ((size_t)b * NN + mrow) * DH;
#pragma unroll
        for (int dt = 0; dt < NT; ++dt) {
            int dcol = dt * 16 + lrow;
            if (dcol < DH) crow[dcol] = acc[dt][r] * inv;
        }
    }
}

// ---------------- 7) c2 = softmax_col(att)^T @ h1 (transposed A reads) ----------------
__global__ void c2_kernel(const float* __restrict__ att,
                          const float* __restrict__ cmax, const float* __restrict__ csum,
                          const _Float16* __restrict__ h1T,
                          float* __restrict__ c2) {
    int wid  = blockIdx.x * 8 + (threadIdx.x >> 5);   // b*32 + mt
    int lane = threadIdx.x & 31;
    int half = lane >> 4, lrow = lane & 15;
    int b = wid >> 5, mt = wid & 31;

    int m_g = mt * 16 + lrow;                          // output row = att column j
    float cmx = cmax[b * NN + m_g];
    const float*    abase = att + (size_t)b * NN * NN + m_g;
    const _Float16* bbase = h1T + ((size_t)b * DPAD + lrow) * NN;

    v8f acc[NT];
#pragma unroll
    for (int t = 0; t < NT; ++t) acc[t] = (v8f){};

    for (int k0 = 0; k0 < NN; k0 += 32) {
        int ka = k0 + half * 16;
        v16h a;
#pragma unroll
        for (int i = 0; i < 8; ++i)
            a[i]     = (_Float16)__expf(abase[(size_t)(ka + i) * NN] - cmx);
#pragma unroll
        for (int i = 0; i < 8; ++i)
            a[8 + i] = (_Float16)__expf(abase[(size_t)(ka + 8 + i) * NN] - cmx);
#pragma unroll
        for (int dt = 0; dt < NT; ++dt) {
            v16h bv = *(const v16h*)(bbase + (size_t)dt * 16 * NN + k0 + half * 16);
            acc[dt] = __builtin_amdgcn_wmma_f32_16x16x32_f16(false, a, false, bv,
                                                             (short)0, acc[dt], false, false);
        }
    }
#pragma unroll
    for (int r = 0; r < 8; ++r) {
        int mrow = mt * 16 + r + 8 * half;
        float inv = 1.0f / csum[b * NN + mrow];
        float* crow = c2 + ((size_t)b * NN + mrow) * DH;
#pragma unroll
        for (int dt = 0; dt < NT; ++dt) {
            int dcol = dt * 16 + lrow;
            if (dcol < DH) crow[dcol] = acc[dt][r] * inv;
        }
    }
}

// ---------------- 8a) pack X = [h | c] -> f16 [BN, 416] ----------------
__global__ void packx_kernel(const float* __restrict__ h, const float* __restrict__ c,
                             _Float16* __restrict__ xpad) {
    size_t idx = (size_t)blockIdx.x * 256 + threadIdx.x;   // over BN*XKP
    int row = (int)(idx / XKP);
    int k   = (int)(idx % XKP);
    float v = (k < DH) ? h[(size_t)row * DH + k]
                       : ((k < 2 * DH) ? c[(size_t)row * DH + (k - DH)] : 0.f);
    xpad[idx] = (_Float16)v;
}

// ---------------- 8b) pack W_down^T -> f16 [208, 416] ----------------
__global__ void wdt_kernel(const float* __restrict__ Wd, _Float16* __restrict__ wdT) {
    int idx = blockIdx.x * 256 + threadIdx.x;              // over DPAD*XKP
    if (idx >= DPAD * XKP) return;
    int d = idx / XKP, k = idx % XKP;
    float v = (d < DH && k < 2 * DH) ? Wd[(size_t)k * DH + d] : 0.f;
    wdT[idx] = (_Float16)v;
}

// ---------------- 8c) down-proj GEMM (WMMA) + relu + degree table + mean acc ------
__global__ void down_wmma_kernel(const _Float16* __restrict__ xpad,
                                 const _Float16* __restrict__ wdT,
                                 const int* __restrict__ deg,
                                 const float* __restrict__ dtab,
                                 float* __restrict__ hsum) {
    int wid  = blockIdx.x * 8 + (threadIdx.x >> 5);   // m-tile id, 0..2047
    int lane = threadIdx.x & 31;
    int half = lane >> 4, lrow = lane & 15;

    const _Float16* arow  = xpad + ((size_t)wid * 16 + lrow) * XKP;
    const _Float16* bbase = wdT + (size_t)lrow * XKP;

    v8f acc[NT];
#pragma unroll
    for (int t = 0; t < NT; ++t) acc[t] = (v8f){};

    for (int k0 = 0; k0 < XKP; k0 += 32) {
        int ka = k0 + half * 16;
        v16h a = make_a(*(const v8h*)(arow + ka), *(const v8h*)(arow + ka + 8));
#pragma unroll
        for (int dt = 0; dt < NT; ++dt) {
            v16h bv = *(const v16h*)(bbase + (size_t)dt * 16 * XKP + k0 + half * 16);
            acc[dt] = __builtin_amdgcn_wmma_f32_16x16x32_f16(false, a, false, bv,
                                                             (short)0, acc[dt], false, false);
        }
    }
#pragma unroll
    for (int r = 0; r < 8; ++r) {
        int row = wid * 16 + r + 8 * half;
        int b   = row >> 9;
        int dg  = deg[row]; if (dg > 199) dg = 199;
        const float* drow = dtab + (size_t)dg * DH;
#pragma unroll
        for (int dt = 0; dt < NT; ++dt) {
            int dcol = dt * 16 + lrow;
            if (dcol < DH)
                atomicAdd(&hsum[b * DH + dcol], fmaxf(acc[dt][r], 0.f) + drow[dcol]);
        }
    }
}

// ---------------- 9) interaction sums ----------------
__global__ void imean_kernel(const float* __restrict__ inter, float* __restrict__ isum) {
    int idx = blockIdx.x * blockDim.x + threadIdx.x;   // b*6 + j
    if (idx >= BB * 6) return;
    int b = idx / 6, j = idx % 6;
    float s = 0.f;
    for (int n = 0; n < NN; ++n) s += inter[((size_t)b * NN + n) * 6 + j];
    isum[idx] = s;
}

// ---------------- 10) readout MLP, one block per batch ----------------
__global__ void readout_kernel(const float* __restrict__ hsum1, const float* __restrict__ isum1,
                               const float* __restrict__ hsum2, const float* __restrict__ isum2,
                               const float* __restrict__ Wf1, const float* __restrict__ bf1,
                               const float* __restrict__ Wf2, const float* __restrict__ bf2,
                               const float* __restrict__ Wf3, const float* __restrict__ bf3,
                               const float* __restrict__ Wf4, const float* __restrict__ bf4,
                               float* __restrict__ out) {
    __shared__ float x[618];
    __shared__ float y1[400];
    __shared__ float y2[200];
    __shared__ float y3[100];
    __shared__ float red[256];
    int b = blockIdx.x, t = threadIdx.x;
    for (int i = t; i < 206; i += 256) {
        float v1 = (i < DH) ? hsum1[b * DH + i] / 512.0f : isum1[b * 6 + (i - DH)] / 512.0f;
        float v2 = (i < DH) ? hsum2[b * DH + i] / 512.0f : isum2[b * 6 + (i - DH)] / 512.0f;
        x[i] = v1; x[206 + i] = v2; x[412 + i] = v1 - v2;
    }
    __syncthreads();
    for (int c = t; c < 400; c += 256) {
        float a = bf1[c];
        for (int k = 0; k < 618; ++k) a += x[k] * Wf1[k * 400 + c];
        y1[c] = fmaxf(a, 0.f);
    }
    __syncthreads();
    for (int c = t; c < 200; c += 256) {
        float a = bf2[c];
        for (int k = 0; k < 400; ++k) a += y1[k] * Wf2[k * 200 + c];
        y2[c] = fmaxf(a, 0.f);
    }
    __syncthreads();
    for (int c = t; c < 100; c += 256) {
        float a = bf3[c];
        for (int k = 0; k < 200; ++k) a += y2[k] * Wf3[k * 100 + c];
        y3[c] = fmaxf(a, 0.f);
    }
    __syncthreads();
    red[t] = (t < 100) ? y3[t] * Wf4[t] : 0.f;
    __syncthreads();
    for (int s = 128; s > 0; s >>= 1) {
        if (t < s) red[t] += red[t + s];
        __syncthreads();
    }
    if (t == 0) out[b] = red[0] + bf4[0];
}

// ---------------- host launcher ----------------
extern "C" void kernel_launch(void* const* d_in, const int* in_sizes, int n_in,
                              void* d_out, int out_size, void* d_ws, size_t ws_size,
                              hipStream_t stream) {
    const float* atom1  = (const float*)d_in[0];
    const float* atom2  = (const float*)d_in[1];
    const float* crd1   = (const float*)d_in[2];
    const float* crd2   = (const float*)d_in[3];
    const float* ef1    = (const float*)d_in[4];
    const float* ef2    = (const float*)d_in[5];
    const float* int1   = (const float*)d_in[6];
    const float* int2   = (const float*)d_in[7];
    const int*   src1   = (const int*)d_in[8];
    const int*   dst1   = (const int*)d_in[9];
    const int*   src2   = (const int*)d_in[10];
    const int*   dst2   = (const int*)d_in[11];
    const float* W_atom = (const float*)d_in[12];
    const float* W_edge = (const float*)d_in[13];
    const float* W_rbf  = (const float*)d_in[14];
    const float* b_rbf  = (const float*)d_in[15];
    const float* ln_g   = (const float*)d_in[16];
    const float* ln_b   = (const float*)d_in[17];
    const float* W_down = (const float*)d_in[18];
    const float* dtab   = (const float*)d_in[19];
    const float* Wf1    = (const float*)d_in[20];
    const float* bf1    = (const float*)d_in[21];
    const float* Wf2    = (const float*)d_in[22];
    const float* bf2    = (const float*)d_in[23];
    const float* Wf3    = (const float*)d_in[24];
    const float* bf3    = (const float*)d_in[25];
    const float* Wf4    = (const float*)d_in[26];
    const float* bf4    = (const float*)d_in[27];
    float* out = (float*)d_out;

    size_t off = 0;
    auto alloc = [&](size_t bytes) -> void* {
        void* p = (char*)d_ws + off;
        off += (bytes + 255) & ~(size_t)255;
        return p;
    };
    float*    h1     = (float*)alloc((size_t)BN * DH * 4);
    float*    h2     = (float*)alloc((size_t)BN * DH * 4);
    int*      deg1   = (int*)alloc((size_t)BN * 4);
    int*      deg2   = (int*)alloc((size_t)BN * 4);
    _Float16* h1pad  = (_Float16*)alloc((size_t)BN * KPAD * 2);
    _Float16* h2pad  = (_Float16*)alloc((size_t)BN * KPAD * 2);
    _Float16* h1T    = (_Float16*)alloc((size_t)BB * DPAD * NN * 2);
    _Float16* h2T    = (_Float16*)alloc((size_t)BB * DPAD * NN * 2);
    float*    att    = (float*)alloc((size_t)BB * NN * NN * 4);
    float*    rmax   = (float*)alloc((size_t)BN * 4);
    float*    rsum   = (float*)alloc((size_t)BN * 4);
    float*    cmax   = (float*)alloc((size_t)BN * 4);
    float*    csum   = (float*)alloc((size_t)BN * 4);
    float*    c1     = (float*)alloc((size_t)BN * DH * 4);
    float*    c2     = (float*)alloc((size_t)BN * DH * 4);
    _Float16* xpad1  = (_Float16*)alloc((size_t)BN * XKP * 2);
    _Float16* xpad2  = (_Float16*)alloc((size_t)BN * XKP * 2);
    _Float16* wdT    = (_Float16*)alloc((size_t)DPAD * XKP * 2);
    float*    hsum1  = (float*)alloc((size_t)BB * DH * 4);
    float*    hsum2  = (float*)alloc((size_t)BB * DH * 4);
    float*    isum1  = (float*)alloc((size_t)BB * 6 * 4);
    float*    isum2  = (float*)alloc((size_t)BB * 6 * 4);
    (void)in_sizes; (void)n_in; (void)out_size; (void)ws_size;

    hipMemsetAsync(deg1, 0, (size_t)BN * 4, stream);
    hipMemsetAsync(deg2, 0, (size_t)BN * 4, stream);
    hipMemsetAsync(hsum1, 0, (size_t)BB * DH * 4, stream);
    hipMemsetAsync(hsum2, 0, (size_t)BB * DH * 4, stream);
    hipMemsetAsync(isum1, 0, (size_t)BB * 6 * 4, stream);
    hipMemsetAsync(isum2, 0, (size_t)BB * 6 * 4, stream);

    // 1) atom embed + LN
    atom_kernel<<<BN / 8, 256, 0, stream>>>(atom1, W_atom, ln_g, ln_b, h1);
    atom_kernel<<<BN / 8, 256, 0, stream>>>(atom2, W_atom, ln_g, ln_b, h2);

    // 2) edge messages (scatter into h, degree counts)
    edge_kernel<<<EE / 8, 256, 0, stream>>>(ef1, crd1, src1, dst1, W_edge, W_rbf, b_rbf,
                                            ln_g, ln_b, h1, deg1);
    edge_kernel<<<EE / 8, 256, 0, stream>>>(ef2, crd2, src2, dst2, W_edge, W_rbf, b_rbf,
                                            ln_g, ln_b, h2, deg2);

    // 3) pack f16
    pack_kernel<<<(BN * KPAD) / 256, 256, 0, stream>>>(h1, h1pad, h1T);
    pack_kernel<<<(BN * KPAD) / 256, 256, 0, stream>>>(h2, h2pad, h2T);

    // 4) attention logits (WMMA)
    att_kernel<<<dim3(8, 8, BB), 256, 0, stream>>>(h1pad, h2pad, att);

    // 5) softmax stats
    rowstats_kernel<<<BN / 8, 256, 0, stream>>>(att, rmax, rsum);
    colstats_kernel<<<BN / 8, 256, 0, stream>>>(att, cmax, csum);

    // 6/7) attention-weighted context (WMMA, register-blocked over 13 N tiles)
    c1_kernel<<<(BB * 32) / 8, 256, 0, stream>>>(att, rmax, rsum, h2T, c1);
    c2_kernel<<<(BB * 32) / 8, 256, 0, stream>>>(att, cmax, csum, h1T, c2);

    // 8) down-projection (WMMA) + degree embedding + batch means
    packx_kernel<<<(BN * XKP) / 256, 256, 0, stream>>>(h1, c1, xpad1);
    packx_kernel<<<(BN * XKP) / 256, 256, 0, stream>>>(h2, c2, xpad2);
    wdt_kernel<<<(DPAD * XKP + 255) / 256, 256, 0, stream>>>(W_down, wdT);
    down_wmma_kernel<<<(BN / 16) / 8, 256, 0, stream>>>(xpad1, wdT, deg1, dtab, hsum1);
    down_wmma_kernel<<<(BN / 16) / 8, 256, 0, stream>>>(xpad2, wdT, deg2, dtab, hsum2);

    // 9) interaction means
    imean_kernel<<<2, 192, 0, stream>>>(int1, isum1);
    imean_kernel<<<2, 192, 0, stream>>>(int2, isum2);

    // 10) readout MLP
    readout_kernel<<<BB, 256, 0, stream>>>(hsum1, isum1, hsum2, isum2,
                                           Wf1, bf1, Wf2, bf2, Wf3, bf3, Wf4, bf4, out);
}
